// LSTMlayer_26560077758826
// MI455X (gfx1250) — compile-verified, matrix operand order checked by reading, not run on hardware
//
#include <hip/hip_runtime.h>

// ---------------------------------------------------------------------------
// LSTM layer for MI455X (gfx1250, wave32, WMMA), B=32 T=2048 D=H=256.
//   cvt_x : x f32 -> bf16 row-major                       (A fragments)
//   cvt_w : W f32 -> bf16 in WMMA B-fragment order        (32B/lane/frag)
//   xw_gemm: xW[t][b][4H] = x@W + 0 (bias added later), 4 N-tiles per wave,
//            fully unrolled K, 32 v_wmma per wave.
//   lstm_rec: 16 persistent WGs, 16 units each; U slice swizzled into LDS in
//            fragment order (ds_load_b128), h double-buffered in global (L2),
//            pointwise adds xW + bias, per-step release/acquire counter sync.
// ---------------------------------------------------------------------------

typedef __attribute__((ext_vector_type(16))) __bf16 v16bf;
typedef __attribute__((ext_vector_type(8)))  __bf16 v8bf;
typedef __attribute__((ext_vector_type(4)))  __bf16 v4bf;
typedef __attribute__((ext_vector_type(2)))  __bf16 v2bf;
typedef __attribute__((ext_vector_type(8)))  float  v8f;
typedef __attribute__((ext_vector_type(4)))  float  v4f;
typedef __attribute__((ext_vector_type(2)))  float  v2f;

#define BATCH 32
#define SEQT  2048
#define DIN   256
#define HU    256
#define G4    1024                 // 4*HU
#define NROW  (BATCH * SEQT)       // 65536
#define NWG   16
#define UPW   (HU / NWG)           // 16

__device__ __forceinline__ float sigf(float v) { return 1.0f / (1.0f + __expf(-v)); }

// -------------------------- init: zero h + counters -------------------------
__global__ void lstm_init(__bf16* __restrict__ hbuf, int* __restrict__ cnt) {
    int i = blockIdx.x * blockDim.x + threadIdx.x;
    if (i < 2 * BATCH * HU) hbuf[i] = (__bf16)0.0f;
    if (i < SEQT)           cnt[i]  = 0;
}

// -------------------------- x -> bf16 (row major) ---------------------------
__global__ void cvt_x(const float* __restrict__ x, __bf16* __restrict__ xbf) {
    long i = ((long)blockIdx.x * blockDim.x + threadIdx.x) * 4;   // NROW*DIN elems
    v4f v = *(const v4f*)(x + i);
    v4bf o;
#pragma unroll
    for (int e = 0; e < 4; ++e) o[e] = (__bf16)v[e];
    *(v4bf*)(xbf + i) = o;
}

// ------------------ W -> bf16 in WMMA B-fragment order ----------------------
// Wfrag[nt][kb][lane][e], nt<64, kb<8, lane<32, e<16  (32B per lane-fragment)
__global__ void cvt_w(const float* __restrict__ W, __bf16* __restrict__ Wfrag) {
    int tg    = blockIdx.x * 256 + threadIdx.x;     // < 64*8*32 = 16384
    int nt    = tg >> 8;
    int kb    = (tg >> 5) & 7;
    int ln    = tg & 31;
    int khalf = (ln >> 4) << 3;
    int n     = nt * 16 + (ln & 15);
    v16bf o;
#pragma unroll
    for (int e = 0; e < 8; ++e) {
        o[e]     = (__bf16)W[(long)(kb * 32 + khalf + e) * G4 + n];
        o[e + 8] = (__bf16)W[(long)(kb * 32 + 16 + khalf + e) * G4 + n];
    }
    *(v16bf*)(Wfrag + (long)tg * 16) = o;
}

// -------------------------- phase 1: xW = x@W -------------------------------
// Wave task: one 16-row tile x 4 col-tiles (A reused 4x). xW layout [T][B][G4].
__global__ __launch_bounds__(256) void xw_gemm(const __bf16* __restrict__ xbf,
                                               const __bf16* __restrict__ Wfrag,
                                               float* __restrict__ xW) {
    const int wave  = threadIdx.x >> 5;
    const int lane  = threadIdx.x & 31;
    const int task  = blockIdx.x * 8 + wave;        // < 4096*16
    const int mt    = task >> 4;                    // row tile
    const int ng    = task & 15;                    // group of 4 col tiles
    const int nloc  = lane & 15;
    const int half  = lane >> 4;
    const int khalf = half << 3;
    const int arow  = mt * 16 + nloc;

    v8f acc[4] = {};
#pragma unroll
    for (int kb = 0; kb < 8; ++kb) {
        const __bf16* ap = xbf + (long)arow * DIN + kb * 32 + khalf;
        v8bf alo = *(const v8bf*)ap;
        v8bf ahi = *(const v8bf*)(ap + 16);
        v16bf a = __builtin_shufflevector(alo, ahi, 0, 1, 2, 3, 4, 5, 6, 7,
                                          8, 9, 10, 11, 12, 13, 14, 15);
#pragma unroll
        for (int j = 0; j < 4; ++j) {
            v16bf b = *(const v16bf*)(Wfrag + ((long)((ng * 4 + j) * 8 + kb) * 32 + lane) * 16);
            acc[j] = __builtin_amdgcn_wmma_f32_16x16x32_bf16(false, a, false, b,
                                                             (short)0, acc[j], false, false);
        }
    }
    // store to xW[t][b][col]; 16-row tile stays inside one batch (T%16==0)
    const int b_ = (mt * 16) >> 11;       // row / SEQT
    const int t0 = (mt * 16) & (SEQT - 1);
#pragma unroll
    for (int j = 0; j < 4; ++j) {
        int col = (ng * 4 + j) * 16 + nloc;
#pragma unroll
        for (int r = 0; r < 8; ++r) {
            int mrow = r + (half << 3);
            xW[((long)(t0 + mrow) * BATCH + b_) * G4 + col] = acc[j][r];
        }
    }
}

// -------------------------- phase 2: recurrence -----------------------------
__global__ __launch_bounds__(256, 1) void lstm_rec(const float* __restrict__ U,
                                                   const float* __restrict__ xW,
                                                   const float* __restrict__ bias,
                                                   __bf16* hbuf,
                                                   int* cnt,
                                                   float* __restrict__ out) {
    // U slice in WMMA B-fragment order: Ub[g][kb][lane][e] (32B/lane-fragment)
    __shared__ __attribute__((aligned(32))) __bf16 Ub[4 * 8 * 32 * 16];   // 32 KB
    __shared__ __attribute__((aligned(16))) float  gb[4 * BATCH * UPW];   // 8 KB
    __shared__ __attribute__((aligned(16))) float  cs[BATCH * UPW];       // 2 KB

    const int wg   = blockIdx.x;
    const int tid  = threadIdx.x;
    const int lane = tid & 31;
    const int wave = tid >> 5;

    // ---- preload + swizzle U columns {g*HU + wg*16 + n} into fragment order
    for (int idx = tid; idx < 4 * 8 * 32; idx += 256) {
        int g     = idx >> 8;
        int kb    = (idx >> 5) & 7;
        int ln    = idx & 31;
        int khalf = (ln >> 4) << 3;
        int n     = ln & 15;
        v16bf o;
#pragma unroll
        for (int e = 0; e < 8; ++e) {
            o[e]     = (__bf16)U[(long)(kb * 32 + khalf + e) * G4 + g * HU + wg * UPW + n];
            o[e + 8] = (__bf16)U[(long)(kb * 32 + 16 + khalf + e) * G4 + g * HU + wg * UPW + n];
        }
        *(v16bf*)(Ub + idx * 16) = o;
    }
    for (int idx = tid; idx < BATCH * UPW; idx += 256) cs[idx] = 0.0f;

    // pointwise-stage constants: each thread owns (batch b_, unit pair 2*np)
    const int b_ = tid >> 3;            // 0..31
    const int nn = (tid & 7) << 1;      // 0,2,..,14
    v2f bi[4];
#pragma unroll
    for (int g = 0; g < 4; ++g) bi[g] = *(const v2f*)&bias[g * HU + wg * UPW + nn];
    __syncthreads();

    const int mt    = wave >> 2;        // batch-half
    const int g     = wave & 3;         // gate
    const int nloc  = lane & 15;
    const int half  = lane >> 4;
    const int khalf = half << 3;
    const int arow  = mt * 16 + nloc;

#pragma unroll 1
    for (int t = 0; t < SEQT; ++t) {
        const __bf16* hcur = hbuf + (size_t)(t & 1) * (BATCH * HU);
        __bf16*       hnxt = hbuf + (size_t)((t + 1) & 1) * (BATCH * HU);

        // ---- h(t) @ U_slice : 8 chained WMMAs over K=256 ----
        v8f acc = {};
#pragma unroll
        for (int kb = 0; kb < 8; ++kb) {
            const __bf16* ap = hcur + (long)arow * HU + kb * 32 + khalf;
            v8bf alo = *(const v8bf*)ap;
            v8bf ahi = *(const v8bf*)(ap + 16);
            v16bf a = __builtin_shufflevector(alo, ahi, 0, 1, 2, 3, 4, 5, 6, 7,
                                              8, 9, 10, 11, 12, 13, 14, 15);
            v16bf b = *(const v16bf*)(Ub + ((g * 8 + kb) * 32 + lane) * 16);
            acc = __builtin_amdgcn_wmma_f32_16x16x32_bf16(false, a, false, b,
                                                          (short)0, acc, false, false);
        }
#pragma unroll
        for (int r = 0; r < 8; ++r) {
            int brow = mt * 16 + r + (half << 3);
            gb[(g * BATCH + brow) * UPW + nloc] = acc[r];
        }
        __syncthreads();

        // ---- pointwise cell update: thread -> (b_, units nn..nn+1) ----
        {
            const long xb = ((long)t * BATCH + b_) * G4 + wg * UPW + nn;
            v2f gi = *(const v2f*)&gb[(0 * BATCH + b_) * UPW + nn];
            v2f gf = *(const v2f*)&gb[(1 * BATCH + b_) * UPW + nn];
            v2f gg = *(const v2f*)&gb[(2 * BATCH + b_) * UPW + nn];
            v2f go = *(const v2f*)&gb[(3 * BATCH + b_) * UPW + nn];
            gi += *(const v2f*)&xW[xb + 0 * HU] + bi[0];
            gf += *(const v2f*)&xW[xb + 1 * HU] + bi[1];
            gg += *(const v2f*)&xW[xb + 2 * HU] + bi[2];
            go += *(const v2f*)&xW[xb + 3 * HU] + bi[3];
            if (t + 1 < SEQT) {     // prefetch next timestep's slab (L2)
                const long xn = ((long)(t + 1) * BATCH + b_) * G4 + wg * UPW + nn;
#pragma unroll
                for (int pg = 0; pg < 4; ++pg)
                    __builtin_prefetch(&xW[xn + pg * HU], 0, 0);
            }
            v2f cp = *(const v2f*)&cs[b_ * UPW + nn];
            v2f cN, hN;
#pragma unroll
            for (int e = 0; e < 2; ++e) {
                float i_ = sigf(gi[e]);
                float f_ = sigf(gf[e]);
                float g_ = tanhf(gg[e]);
                float o_ = sigf(go[e]);
                cN[e] = f_ * cp[e] + i_ * g_;
                hN[e] = o_ * tanhf(cN[e]);
            }
            *(v2f*)&cs[b_ * UPW + nn] = cN;
            *(v2f*)(out + ((long)b_ * SEQT + t) * HU + wg * UPW + nn) = hN;
            v2bf hb;
            hb[0] = (__bf16)hN[0];
            hb[1] = (__bf16)hN[1];
            *(v2bf*)(hnxt + (long)b_ * HU + wg * UPW + nn) = hb;
        }
        __threadfence();
        __syncthreads();

        // ---- cross-WG sync: all 16 slices of h(t+1) published ----
        if (tid == 0) {
            __hip_atomic_fetch_add(&cnt[t], 1, __ATOMIC_RELEASE,
                                   __HIP_MEMORY_SCOPE_AGENT);
            while (__hip_atomic_load(&cnt[t], __ATOMIC_ACQUIRE,
                                     __HIP_MEMORY_SCOPE_AGENT) < NWG) {
                __builtin_amdgcn_s_sleep(1);
            }
        }
        __syncthreads();
    }
}

// ---------------------------------------------------------------------------
extern "C" void kernel_launch(void* const* d_in, const int* in_sizes, int n_in,
                              void* d_out, int out_size, void* d_ws, size_t ws_size,
                              hipStream_t stream) {
    (void)in_sizes; (void)n_in; (void)out_size; (void)ws_size;
    const float* x   = (const float*)d_in[0];   // [B,T,D]
    const float* W   = (const float*)d_in[1];   // [D,4H]
    const float* U   = (const float*)d_in[2];   // [H,4H]
    const float* b   = (const float*)d_in[3];   // [4H]
    float*       out = (float*)d_out;           // [B,T,H]

    // workspace: xW f32 [T][B][4H] (256MB) | xbf (32MB) | Wfrag (512KB)
    //            | h double buffer bf16 (32KB) | counters (8KB)
    char*   p     = (char*)d_ws;
    float*  xW    = (float*)p;            p += (size_t)SEQT * BATCH * G4 * sizeof(float);
    __bf16* xbf   = (__bf16*)p;           p += (size_t)NROW * DIN * sizeof(__bf16);
    __bf16* Wfrag = (__bf16*)p;           p += (size_t)64 * 8 * 32 * 16 * sizeof(__bf16);
    __bf16* hbuf  = (__bf16*)p;           p += (size_t)2 * BATCH * HU * sizeof(__bf16);
    int*    cnt   = (int*)p;

    lstm_init<<<64, 256, 0, stream>>>(hbuf, cnt);
    cvt_x<<<(NROW * DIN / 4) / 256, 256, 0, stream>>>(x, xbf);
    cvt_w<<<64, 256, 0, stream>>>(W, Wfrag);
    xw_gemm<<<(NROW / 16) * (G4 / 64) / 8, 256, 0, stream>>>(xbf, Wfrag, xW);
    lstm_rec<<<NWG, 256, 0, stream>>>(U, xW, b, hbuf, cnt, out);
}
